// AttentionCellEncoder_27711128994330
// MI455X (gfx1250) — matrix operand order repaired and checked by reading, not compiled, not run on hardware
//
#include <hip/hip_runtime.h>

// ---------------- problem constants ----------------
#define D_ 768
#define H_ 512
#define O_ 256
#define NH_ 8
#define HD_ 64
#define C_ 4096
#define L_ 32

// ---------------- vector types (POD ext-vectors, union-safe) ----------------
typedef __bf16        v16bf __attribute__((ext_vector_type(16)));
typedef float         v8f   __attribute__((ext_vector_type(8)));
typedef unsigned int  u32x4 __attribute__((ext_vector_type(4)));
typedef float         f32x4 __attribute__((ext_vector_type(4)));

union Frag { v16bf v; u32x4 q[2]; };

// fp32 -> bf16 round-to-nearest-even
__device__ __forceinline__ unsigned short f2bf(float f) {
  unsigned u = __float_as_uint(f);
  unsigned r = u + 0x7FFFu + ((u >> 16) & 1u);
  return (unsigned short)(r >> 16);
}

// Packed B-fragment layout this file uses for V_WMMA_F32_16X16X32_BF16 (contiguous k halves, same order as A).
// B is K(32) x N(16) per tile. VGPR v of lane l holds packed {K=2v,K=2v+1}
// (lanes 0-15, K=0..15) or {K=2v+16,K=2v+17} (lanes 16-31), N = l&15.
// Tile id = nt*NKT + kt ; per tile 32 lanes x 16 bf16 contiguous.
__device__ __forceinline__ int packIdxB(int n, int k, int nkt) {
  int nt = n >> 4, kt = k >> 5;
  int lane = (n & 15) | (((k >> 4) & 1) << 4);
  int e = k & 15;
  return ((nt * nkt + kt) * 512) + lane * 16 + e;
}

__device__ __forceinline__ v8f wmma_bf16(const Frag& a, const Frag& b, v8f c) {
  return __builtin_amdgcn_wmma_f32_16x16x32_bf16(false, a.v, false, b.v,
                                                 (short)0, c, false, false);
}

// ---------------- workspace layout (bytes) ----------------
#define WF_OFF   0u           // bf16 fused QKV weights, 3 * 768*512 * 2B
#define WF_PSZ   (768*512)    // elements per projection
#define BF_OFF   2359296u     // fp32 fused biases, 3*512
#define WFIN_OFF 2365440u     // fp32 WfinalT (transposed Wout@Wo), 512*256
#define BFIN_OFF 2889728u     // fp32 bfinal, 256

// =====================================================================
// Prep 1: Wf_p = Wi_p @ W_p  (512x768 bf16, pre-swizzled B-fragment order)
// grid = 3*64 blocks, 256 threads. Block: p, 8 rows of n, full k (3 cols/thr)
// =====================================================================
__global__ void fuse_qkv_kernel(const float* __restrict__ in_w,
                                const float* __restrict__ Wq,
                                const float* __restrict__ Wk,
                                const float* __restrict__ Wv,
                                unsigned short* __restrict__ wf) {
  int b = blockIdx.x;
  int p = b >> 6;
  int n0 = (b & 63) * 8;
  const float* Wp = (p == 0) ? Wq : ((p == 1) ? Wk : Wv);
  __shared__ float wi[8 * 512];
  for (int i = threadIdx.x; i < 8 * 512; i += 256) {
    int j = i >> 9, h = i & 511;
    wi[i] = in_w[(size_t)(p * 512 + n0 + j) * 512 + h];
  }
  __syncthreads();
  float acc[8][3];
#pragma unroll
  for (int j = 0; j < 8; ++j)
#pragma unroll
    for (int m = 0; m < 3; ++m) acc[j][m] = 0.f;
  int k0 = threadIdx.x;
  for (int h = 0; h < 512; ++h) {
    float w0 = Wp[(size_t)h * 768 + k0];
    float w1 = Wp[(size_t)h * 768 + k0 + 256];
    float w2 = Wp[(size_t)h * 768 + k0 + 512];
#pragma unroll
    for (int j = 0; j < 8; ++j) {
      float a = wi[j * 512 + h];
      acc[j][0] += a * w0; acc[j][1] += a * w1; acc[j][2] += a * w2;
    }
  }
#pragma unroll
  for (int j = 0; j < 8; ++j)
#pragma unroll
    for (int m = 0; m < 3; ++m) {
      int n = n0 + j, k = k0 + m * 256;
      wf[(size_t)p * WF_PSZ + packIdxB(n, k, 24)] = f2bf(acc[j][m]);
    }
}

// =====================================================================
// Prep 2: WfinalT[k][o] = (Wout @ Wo)[o][k]  (512x256 fp32, TRANSPOSED so the
// main kernel's final mat-vec reads are lane-coalesced).
// grid = 32 blocks, 256 thr.
// =====================================================================
__global__ void fuse_final_kernel(const float* __restrict__ Wout,
                                  const float* __restrict__ Wo,
                                  float* __restrict__ wfinT) {
  int n0 = blockIdx.x * 8;
  __shared__ float wo[8 * 512];
  for (int i = threadIdx.x; i < 8 * 512; i += 256) {
    int j = i >> 9, h = i & 511;
    wo[i] = Wout[(size_t)(n0 + j) * 512 + h];
  }
  __syncthreads();
  float acc[8][2];
#pragma unroll
  for (int j = 0; j < 8; ++j) { acc[j][0] = 0.f; acc[j][1] = 0.f; }
  int k0 = threadIdx.x;
  for (int h = 0; h < 512; ++h) {
    float w0 = Wo[(size_t)h * 512 + k0];
    float w1 = Wo[(size_t)h * 512 + k0 + 256];
#pragma unroll
    for (int j = 0; j < 8; ++j) {
      float a = wo[j * 512 + h];
      acc[j][0] += a * w0; acc[j][1] += a * w1;
    }
  }
#pragma unroll
  for (int j = 0; j < 8; ++j) {
    wfinT[(size_t)(k0)       * 256 + n0 + j] = acc[j][0];
    wfinT[(size_t)(k0 + 256) * 256 + n0 + j] = acc[j][1];
  }
}

// =====================================================================
// Prep 3: fused biases. bf_p = Wi_p@b_p + bi_p ; bfinal = Wout@bo + bout
// grid = 7 blocks * 256 = 1792 threads
// =====================================================================
__global__ void fuse_bias_kernel(const float* __restrict__ in_w,
                                 const float* __restrict__ in_b,
                                 const float* __restrict__ bq,
                                 const float* __restrict__ bk,
                                 const float* __restrict__ bv,
                                 const float* __restrict__ Wout,
                                 const float* __restrict__ bo,
                                 const float* __restrict__ bout,
                                 float* __restrict__ bf,
                                 float* __restrict__ bfin) {
  int id = blockIdx.x * 256 + threadIdx.x;
  if (id < 1536) {
    int p = id / 512;
    const float* bp = (p == 0) ? bq : ((p == 1) ? bk : bv);
    float acc = in_b[id];
    for (int h = 0; h < 512; ++h) acc += in_w[(size_t)id * 512 + h] * bp[h];
    bf[id] = acc;
  } else if (id < 1792) {
    int o = id - 1536;
    float acc = bout[o];
    for (int h = 0; h < 512; ++h) acc += Wout[(size_t)o * 512 + h] * bo[h];
    bfin[o] = acc;
  }
}

// =====================================================================
// Main: one workgroup per cell. 256 threads = 8 waves (wave32).
// =====================================================================
__global__ __launch_bounds__(256)
void cell_attention_kernel(const float* __restrict__ chunk,
                           const int* __restrict__ cell_indices,
                           const int* __restrict__ cell_lengths,
                           const unsigned short* __restrict__ wf,
                           const float* __restrict__ bf,
                           const float* __restrict__ wfinT,
                           const float* __restrict__ bfin,
                           float* __restrict__ out) {
  __shared__ __align__(32) unsigned short sh_q[32 * 512];   // 32 KB
  __shared__ __align__(32) unsigned short sh_k[32 * 512];   // 32 KB
  __shared__ __align__(32) unsigned short sh_vT[512 * 32];  // 32 KB (col major)
  __shared__ __align__(32) union {
    unsigned short x[32 * 768];                             // 48 KB (gathered x)
    struct { unsigned short attn[NH_ * 32 * 32]; float pooled[512]; } b;
  } sh_u;
  __shared__ int sh_idx[32];
  __shared__ int sh_len;

  const int cell = blockIdx.x;
  const int tid  = threadIdx.x;
  const int wave = tid >> 5;
  const int lane = tid & 31;
  const int hw   = lane >> 4;   // half-wave id (0/1)
  const int l15  = lane & 15;

  if (tid < 32) sh_idx[tid] = cell_indices[cell * L_ + tid];
  if (tid == 0) {
    int l = cell_lengths[cell];
    sh_len = l < 1 ? 1 : (l > 32 ? 32 : l);
  }
  __syncthreads();

  // ---- gather x (32 x 768 fp32) -> bf16 LDS ----
#pragma unroll
  for (int i = 0; i < 24; ++i) {
    int linear = tid + i * 256;            // 6144 float4 slots
    int row = linear / 192, c4 = linear % 192;
    const f32x4 f = ((const f32x4*)(chunk + (size_t)sh_idx[row] * D_))[c4];
    unsigned int u0 = (unsigned)f2bf(f.x) | ((unsigned)f2bf(f.y) << 16);
    unsigned int u1 = (unsigned)f2bf(f.z) | ((unsigned)f2bf(f.w) << 16);
    ((unsigned int*)sh_u.x)[linear * 2 + 0] = u0;
    ((unsigned int*)sh_u.x)[linear * 2 + 1] = u1;
  }
  __syncthreads();

  // ---- fused QKV projections: (32x768) @ (768x512) per p ----
  // Software-pipelined K loop: ping-pong A (LDS) and B (global) fragment
  // buffers so next-step loads overlap current-step WMMAs.
  const unsigned short* xbase = sh_u.x;
  const float qscale = 0.125f;  // 1/sqrt(HD=64)

#define LOAD_A(dst, kt)                                                        \
  {                                                                            \
    _Pragma("unroll") for (int mt = 0; mt < 2; ++mt) {                         \
      const unsigned short* ap = &xbase[(mt * 16 + l15) * D_ + (kt) * 32 + 16 * hw]; \
      dst[mt].q[0] = *(const u32x4*)(ap);                                      \
      dst[mt].q[1] = *(const u32x4*)(ap + 8);                                  \
    }                                                                          \
  }
#define LOAD_B(dst, pbase, kt)                                                 \
  {                                                                            \
    _Pragma("unroll") for (int j = 0; j < 4; ++j) {                            \
      const u32x4* bp = (const u32x4*)((pbase) +                               \
          (((wave * 4 + j) * 24 + (kt)) * 512 + lane * 16));                   \
      dst[j].q[0] = bp[0];                                                     \
      dst[j].q[1] = bp[1];                                                     \
    }                                                                          \
  }
#define MMA8(acc, a, b)                                                        \
  {                                                                            \
    _Pragma("unroll") for (int j = 0; j < 4; ++j) {                            \
      acc[0][j] = wmma_bf16(a[0], b[j], acc[0][j]);                            \
      acc[1][j] = wmma_bf16(a[1], b[j], acc[1][j]);                            \
    }                                                                          \
  }

  for (int p = 0; p < 3; ++p) {
    const unsigned short* pbase = wf + (size_t)p * WF_PSZ;
    v8f acc[2][4] = {};
    Frag aE[2], bE[4], aO[2], bO[4];
    LOAD_A(aE, 0);
    LOAD_B(bE, pbase, 0);
    for (int kt = 0; kt < 24; kt += 2) {
      LOAD_A(aO, kt + 1);
      LOAD_B(bO, pbase, kt + 1);
      MMA8(acc, aE, bE);
      if (kt + 2 < 24) {
        LOAD_A(aE, kt + 2);
        LOAD_B(bE, pbase, kt + 2);
      }
      MMA8(acc, aO, bO);
    }
    // epilogue: +bias, (q: *scale), bf16, store
#pragma unroll
    for (int j = 0; j < 4; ++j) {
      int n = (wave * 4 + j) * 16 + l15;
      float bias = bf[p * 512 + n];
#pragma unroll
      for (int mt = 0; mt < 2; ++mt)
#pragma unroll
        for (int r = 0; r < 8; ++r) {
          int M = mt * 16 + r + 8 * hw;
          float val = acc[mt][j][r] + bias;
          if (p == 0) val *= qscale;
          unsigned short bv = f2bf(val);
          if (p == 0)      sh_q[M * H_ + n] = bv;
          else if (p == 1) sh_k[M * H_ + n] = bv;
          else             sh_vT[n * 32 + M] = bv;
        }
    }
  }
  __syncthreads();

  // ---- attention: wave == head ----
  const int h = wave;
  const int len = sh_len;
  {
    v8f sacc[2][2] = {};
#pragma unroll
    for (int ks = 0; ks < 2; ++ks) {
      Frag a[2], b[2];
#pragma unroll
      for (int mt = 0; mt < 2; ++mt) {
        const unsigned short* ap = &sh_q[(mt * 16 + l15) * H_ + h * HD_ + ks * 32 + 16 * hw];
        a[mt].q[0] = *(const u32x4*)(ap);
        a[mt].q[1] = *(const u32x4*)(ap + 8);
      }
#pragma unroll
      for (int nt = 0; nt < 2; ++nt) {
        // B[k][n] = k_bf[n][k]: lane reads 16 contiguous bf16 of row n
        b[nt].v = *(const v16bf*)&sh_k[(nt * 16 + l15) * H_ + h * HD_ + ks * 32 + 16 * hw];
      }
#pragma unroll
      for (int mt = 0; mt < 2; ++mt)
#pragma unroll
        for (int nt = 0; nt < 2; ++nt)
          sacc[mt][nt] = wmma_bf16(a[mt], b[nt], sacc[mt][nt]);
    }
    // softmax per row (16-lane group holds one row across 2 N-tiles)
#pragma unroll
    for (int mt = 0; mt < 2; ++mt)
#pragma unroll
      for (int r = 0; r < 8; ++r) {
        int M = mt * 16 + r + 8 * hw;
        int c0 = l15, c1 = 16 + l15;
        float v0 = (c0 < len) ? sacc[mt][0][r] : -1e30f;
        float v1 = (c1 < len) ? sacc[mt][1][r] : -1e30f;
        float mx = fmaxf(v0, v1);
#pragma unroll
        for (int off = 8; off > 0; off >>= 1)
          mx = fmaxf(mx, __shfl_xor(mx, off, 16));
        float e0 = __expf(v0 - mx), e1 = __expf(v1 - mx);
        float s = e0 + e1;
#pragma unroll
        for (int off = 8; off > 0; off >>= 1) s += __shfl_xor(s, off, 16);
        float inv = 1.0f / s;
        sh_u.b.attn[h * 1024 + M * 32 + c0] = f2bf(e0 * inv);
        sh_u.b.attn[h * 1024 + M * 32 + c1] = f2bf(e1 * inv);
      }
  }
  __syncthreads();

  // ---- ctx = attn @ v ; masked pooled row-sum (no atomics needed) ----
  {
    v8f cacc[2][4] = {};
    Frag a[2];
#pragma unroll
    for (int mt = 0; mt < 2; ++mt) {
      const unsigned short* ap = &sh_u.b.attn[h * 1024 + (mt * 16 + l15) * 32 + 16 * hw];
      a[mt].q[0] = *(const u32x4*)(ap);
      a[mt].q[1] = *(const u32x4*)(ap + 8);
    }
#pragma unroll
    for (int nt = 0; nt < 4; ++nt) {
      Frag b;
      b.v = *(const v16bf*)&sh_vT[(h * HD_ + nt * 16 + l15) * 32 + 16 * hw];
      cacc[0][nt] = wmma_bf16(a[0], b, cacc[0][nt]);
      cacc[1][nt] = wmma_bf16(a[1], b, cacc[1][nt]);
    }
#pragma unroll
    for (int nt = 0; nt < 4; ++nt) {
      float s = 0.f;
#pragma unroll
      for (int mt = 0; mt < 2; ++mt)
#pragma unroll
        for (int r = 0; r < 8; ++r) {
          int M = mt * 16 + r + 8 * hw;
          if (M < len) s += cacc[mt][nt][r];
        }
      s += __shfl_xor(s, 16, 32);  // combine the two M-half partials (same n)
      if (lane < 16) sh_u.b.pooled[h * HD_ + nt * 16 + l15] = s;
    }
  }
  __syncthreads();

  // ---- out[o] = (pooled/len) @ Wfinal[o] + bfinal[o]  (coalesced: WfinalT) --
  {
    int o = tid;
    float acc = 0.f;
#pragma unroll 8
    for (int k = 0; k < 512; ++k) {
      acc += sh_u.b.pooled[k] * wfinT[(size_t)k * 256 + o];
    }
    float invlen = 1.0f / (float)len;
    out[(size_t)cell * O_ + o] = acc * invlen + bfin[o];
  }
}

// =====================================================================
extern "C" void kernel_launch(void* const* d_in, const int* in_sizes, int n_in,
                              void* d_out, int out_size, void* d_ws, size_t ws_size,
                              hipStream_t stream) {
  const float* chunk = (const float*)d_in[0];
  const float* Wq    = (const float*)d_in[1];
  const float* bq    = (const float*)d_in[2];
  const float* Wk    = (const float*)d_in[3];
  const float* bk    = (const float*)d_in[4];
  const float* Wv    = (const float*)d_in[5];
  const float* bv    = (const float*)d_in[6];
  const float* in_w  = (const float*)d_in[7];
  const float* in_b  = (const float*)d_in[8];
  const float* Wo    = (const float*)d_in[9];
  const float* bo    = (const float*)d_in[10];
  const float* Wout  = (const float*)d_in[11];
  const float* bout  = (const float*)d_in[12];
  const int* cell_indices = (const int*)d_in[13];
  const int* cell_lengths = (const int*)d_in[14];
  float* out = (float*)d_out;

  char* ws = (char*)d_ws;
  unsigned short* wf = (unsigned short*)(ws + WF_OFF);
  float* bf    = (float*)(ws + BF_OFF);
  float* wfinT = (float*)(ws + WFIN_OFF);
  float* bfin  = (float*)(ws + BFIN_OFF);

  fuse_qkv_kernel<<<dim3(3 * 64), dim3(256), 0, stream>>>(in_w, Wq, Wk, Wv, wf);
  fuse_final_kernel<<<dim3(32), dim3(256), 0, stream>>>(Wout, Wo, wfinT);
  fuse_bias_kernel<<<dim3(7), dim3(256), 0, stream>>>(in_w, in_b, bq, bk, bv,
                                                      Wout, bo, bout, bf, bfin);
  cell_attention_kernel<<<dim3(C_), dim3(256), 0, stream>>>(
      chunk, cell_indices, cell_lengths, wf, bf, wfinT, bfin, out);
}
